// MixtureOfExperts_54503134986268
// MI455X (gfx1250) — matrix operand order repaired and checked by reading, not compiled, not run on hardware
//
#include <hip/hip_runtime.h>
#include <hip/hip_bf16.h>

// Problem constants (reference: B=8, S=2048, D=1024, H=4096, E=8, k=2)
#define TD  1024            // D
#define TH  4096            // H
#define TE  8               // E
#define TT  16384           // T = B*S
#define MT  64              // token tile rows per expert block (4 row-tiles of 16)
#define HC  1024            // H chunk per phase1/phase2 round (4 chunks)
#define LDKX (TD + 8)       // padded LDS stride for x tile (bf16 elems)
#define LDHC (HC + 8)       // padded LDS stride for h chunk (bf16 elems)

typedef __attribute__((ext_vector_type(16))) __bf16 v16bf;
typedef __attribute__((ext_vector_type(8)))  __bf16 v8bf;
typedef __attribute__((ext_vector_type(4)))  __bf16 v4bf;
typedef __attribute__((ext_vector_type(8)))  float  v8f;

// ---- WMMA fragment loaders --------------------------------------------------
// A (16x32 bf16, MxK): lane L (r=L&15, hi=L>>4) holds row M=r,
//   v0..3 = K = hi*8+0..7 ; v4..7 = K = 16+hi*8+0..7
__device__ __forceinline__ v16bf load_a_frag(const __bf16* row, int k0, int hi) {
    v8bf lo = *(const v8bf*)(row + k0 + hi * 8);
    v8bf hh = *(const v8bf*)(row + k0 + 16 + hi * 8);
    return __builtin_shufflevector(lo, hh, 0,1,2,3,4,5,6,7,8,9,10,11,12,13,14,15);
}
// B (32x16 bf16, KxN) from row-major B^T [N][K]: lane holds col N=L&15,
//   v0..3 = K = k0 + hi*8+0..7 ; v4..7 = K = k0 + 16+hi*8+0..7
__device__ __forceinline__ v16bf load_b_frag(const __bf16* ncolrow, int k0, int hi) {
    const __bf16* p = ncolrow + k0 + hi * 8;
    v8bf lo = *(const v8bf*)(p);
    v8bf hh = *(const v8bf*)(p + 16);
    return __builtin_shufflevector(lo, hh, 0,1,2,3,4,5,6,7,8,9,10,11,12,13,14,15);
}

__device__ __forceinline__ float gelu_exact(float g) {
    return 0.5f * g * (1.0f + erff(g * 0.70710678118654752f));
}

// ---- prep: fp32 [R][C] -> bf16 transposed [C][R], batched over blockIdx.z ---
__global__ __launch_bounds__(256)
void moe_transpose_bf16(const float* __restrict__ src, __bf16* __restrict__ dst,
                        int R, int C) {
    __shared__ float tile[32][33];
    const float* s = src + (size_t)blockIdx.z * R * C;
    __bf16*      d = dst + (size_t)blockIdx.z * R * C;
    int tx = threadIdx.x, ty = threadIdx.y;
    int c0 = blockIdx.x * 32, r0 = blockIdx.y * 32;
    #pragma unroll
    for (int i = 0; i < 32; i += 8)
        tile[ty + i][tx] = s[(size_t)(r0 + ty + i) * C + (c0 + tx)];
    __syncthreads();
    #pragma unroll
    for (int i = 0; i < 32; i += 8)
        d[(size_t)(c0 + ty + i) * R + (r0 + tx)] = (__bf16)tile[tx][ty + i];
}

// ---- gating: one wave32 per token; top-2 + softmax; atomic append ----------
__global__ __launch_bounds__(256)
void moe_gate(const float* __restrict__ x, const float* __restrict__ gw,
              const float* __restrict__ gb, int* __restrict__ cnt,
              int* __restrict__ tokIdx, float* __restrict__ tokWt) {
    int wv = threadIdx.x >> 5, lane = threadIdx.x & 31;
    int t = blockIdx.x * 8 + wv;
    const float* xr = x + (size_t)t * TD;
    float p[TE];
    #pragma unroll
    for (int e = 0; e < TE; ++e) p[e] = 0.f;
    for (int dd = lane; dd < TD; dd += 32) {
        float xv = xr[dd];
        const float* g = gw + dd * TE;
        #pragma unroll
        for (int e = 0; e < TE; ++e) p[e] += xv * g[e];
    }
    #pragma unroll
    for (int e = 0; e < TE; ++e)
        #pragma unroll
        for (int off = 16; off >= 1; off >>= 1)
            p[e] += __shfl_xor(p[e], off, 32);
    if (lane == 0) {
        float g0[TE];
        #pragma unroll
        for (int e = 0; e < TE; ++e) g0[e] = p[e] + gb[e];
        int i0 = 0;
        #pragma unroll
        for (int e = 1; e < TE; ++e) if (g0[e] > g0[i0]) i0 = e;   // ties -> lowest idx
        int i1 = -1;
        #pragma unroll
        for (int e = 0; e < TE; ++e)
            if (e != i0 && (i1 < 0 || g0[e] > g0[i1])) i1 = e;
        float e1 = __expf(g0[i1] - g0[i0]);
        float inv = 1.0f / (1.0f + e1);
        int p0 = atomicAdd(&cnt[i0], 1);
        tokIdx[i0 * TT + p0] = t; tokWt[i0 * TT + p0] = inv;
        int p1 = atomicAdd(&cnt[i1], 1);
        tokIdx[i1 * TT + p1] = t; tokWt[i1 * TT + p1] = e1 * inv;
    }
}

// ---- fused expert FFN: 64-token tile, H chunked in 4, col-tile pairs -------
__global__ __launch_bounds__(256)
void moe_expert_ffn(const float* __restrict__ x,
                    const float* __restrict__ b1, const float* __restrict__ b2,
                    const __bf16* __restrict__ w1T, const __bf16* __restrict__ w2T,
                    const int* __restrict__ cnt, const int* __restrict__ tokIdx,
                    const float* __restrict__ tokWt, float* __restrict__ out) {
    int e = blockIdx.y;
    int n_e = cnt[e];
    if ((int)blockIdx.x * MT >= n_e) return;

    __shared__ __bf16 xs[MT * LDKX];   // 64 x 1024 activations (bf16, padded)
    __shared__ __bf16 hs[MT * LDHC];   // 64 x 1024 hidden chunk (bf16, padded)
    __shared__ int    tok_s[MT];
    __shared__ float  wt_s[MT];

    int tid = threadIdx.x;
    if (tid < MT) {
        int pos = blockIdx.x * MT + tid;
        int t = (pos < n_e) ? tokIdx[e * TT + pos] : -1;
        tok_s[tid] = t;
        wt_s[tid]  = (pos < n_e) ? tokWt[e * TT + pos] : 0.0f;
    }
    __syncthreads();

    // phase 0: gather x rows -> bf16 LDS (4 threads per row, 256 floats each)
    {
        int row = tid >> 2;
        int cb  = (tid & 3) * (TD / 4);
        int t   = tok_s[row];
        #pragma unroll 4
        for (int i = 0; i < TD / 4; i += 4) {
            float4 v;
            if (t >= 0) v = *(const float4*)&x[(size_t)t * TD + cb + i];
            else        v = make_float4(0.f, 0.f, 0.f, 0.f);
            v4bf bv = { (__bf16)v.x, (__bf16)v.y, (__bf16)v.z, (__bf16)v.w };
            *(v4bf*)&xs[row * LDKX + cb + i] = bv;
        }
    }
    __syncthreads();

    int wv = tid >> 5, lane = tid & 31;
    int r = lane & 15, hi = lane >> 4;

    // per-row-tile LDS base rows (hoisted; compiler keeps 4 base registers)
    const __bf16* xrow0 = xs + (0 * 16 + r) * LDKX;
    const __bf16* xrow1 = xs + (1 * 16 + r) * LDKX;
    const __bf16* xrow2 = xs + (2 * 16 + r) * LDKX;
    const __bf16* xrow3 = xs + (3 * 16 + r) * LDKX;
    const __bf16* hrow0 = hs + (0 * 16 + r) * LDHC;
    const __bf16* hrow1 = hs + (1 * 16 + r) * LDHC;
    const __bf16* hrow2 = hs + (2 * 16 + r) * LDHC;
    const __bf16* hrow3 = hs + (3 * 16 + r) * LDHC;

    for (int hc = 0; hc < TH / HC; ++hc) {
        // ---- phase 1: h[:, hc*HC : +HC] = gelu(x @ w1 + b1) ---------------
        // wave owns 8 col-tiles, processed in pairs: each A frag feeds 2 B
        for (int cp = 0; cp < 4; ++cp) {
            int n_loc0 = (wv * 8 + cp * 2) * 16 + r;     // col-tile pair
            int n_loc1 = n_loc0 + 16;
            int n_glb0 = hc * HC + n_loc0;
            int n_glb1 = n_glb0 + 16;
            v8f acc[2][4];
            #pragma unroll
            for (int c = 0; c < 2; ++c)
                #pragma unroll
                for (int rt = 0; rt < 4; ++rt)
                    acc[c][rt] = (v8f){0.f,0.f,0.f,0.f,0.f,0.f,0.f,0.f};
            const __bf16* brow0 = w1T + ((size_t)e * TH + n_glb0) * TD;
            const __bf16* brow1 = w1T + ((size_t)e * TH + n_glb1) * TD;
            __builtin_prefetch(brow0 + (size_t)32 * TD, 0, 3); // next pair panel
            __builtin_prefetch(brow1 + (size_t)32 * TD, 0, 3);
            #pragma unroll 2
            for (int k0 = 0; k0 < TD; k0 += 32) {
                v16bf a0 = load_a_frag(xrow0, k0, hi);
                v16bf a1 = load_a_frag(xrow1, k0, hi);
                v16bf a2 = load_a_frag(xrow2, k0, hi);
                v16bf a3 = load_a_frag(xrow3, k0, hi);
                v16bf b0 = load_b_frag(brow0, k0, hi);
                v16bf b1f = load_b_frag(brow1, k0, hi);
                acc[0][0] = __builtin_amdgcn_wmma_f32_16x16x32_bf16(
                    false, a0, false, b0, (short)0, acc[0][0], false, false);
                acc[0][1] = __builtin_amdgcn_wmma_f32_16x16x32_bf16(
                    false, a1, false, b0, (short)0, acc[0][1], false, false);
                acc[0][2] = __builtin_amdgcn_wmma_f32_16x16x32_bf16(
                    false, a2, false, b0, (short)0, acc[0][2], false, false);
                acc[0][3] = __builtin_amdgcn_wmma_f32_16x16x32_bf16(
                    false, a3, false, b0, (short)0, acc[0][3], false, false);
                acc[1][0] = __builtin_amdgcn_wmma_f32_16x16x32_bf16(
                    false, a0, false, b1f, (short)0, acc[1][0], false, false);
                acc[1][1] = __builtin_amdgcn_wmma_f32_16x16x32_bf16(
                    false, a1, false, b1f, (short)0, acc[1][1], false, false);
                acc[1][2] = __builtin_amdgcn_wmma_f32_16x16x32_bf16(
                    false, a2, false, b1f, (short)0, acc[1][2], false, false);
                acc[1][3] = __builtin_amdgcn_wmma_f32_16x16x32_bf16(
                    false, a3, false, b1f, (short)0, acc[1][3], false, false);
            }
            float bias0 = b1[(size_t)e * TH + n_glb0];
            float bias1 = b1[(size_t)e * TH + n_glb1];
            #pragma unroll
            for (int rt = 0; rt < 4; ++rt)
                #pragma unroll
                for (int v = 0; v < 8; ++v) {
                    int mrow = rt * 16 + v + 8 * hi;
                    hs[mrow * LDHC + n_loc0] = (__bf16)gelu_exact(acc[0][rt][v] + bias0);
                    hs[mrow * LDHC + n_loc1] = (__bf16)gelu_exact(acc[1][rt][v] + bias1);
                }
        }
        __syncthreads();

        // ---- phase 2: partial out += h_chunk @ w2[hc*HC:+HC, :] -----------
        for (int cp = 0; cp < 4; ++cp) {
            int n0 = (wv * 8 + cp * 2) * 16 + r;         // column pair in D
            int n1 = n0 + 16;
            v8f acc[2][4];
            #pragma unroll
            for (int c = 0; c < 2; ++c)
                #pragma unroll
                for (int rt = 0; rt < 4; ++rt)
                    acc[c][rt] = (v8f){0.f,0.f,0.f,0.f,0.f,0.f,0.f,0.f};
            const __bf16* brow0 = w2T + ((size_t)e * TD + n0) * TH + hc * HC;
            const __bf16* brow1 = w2T + ((size_t)e * TD + n1) * TH + hc * HC;
            __builtin_prefetch(brow0 + (size_t)32 * TH, 0, 3); // next pair panel
            __builtin_prefetch(brow1 + (size_t)32 * TH, 0, 3);
            #pragma unroll 2
            for (int k0 = 0; k0 < HC; k0 += 32) {
                v16bf a0 = load_a_frag(hrow0, k0, hi);
                v16bf a1 = load_a_frag(hrow1, k0, hi);
                v16bf a2 = load_a_frag(hrow2, k0, hi);
                v16bf a3 = load_a_frag(hrow3, k0, hi);
                v16bf b0 = load_b_frag(brow0, k0, hi);
                v16bf b1f = load_b_frag(brow1, k0, hi);
                acc[0][0] = __builtin_amdgcn_wmma_f32_16x16x32_bf16(
                    false, a0, false, b0, (short)0, acc[0][0], false, false);
                acc[0][1] = __builtin_amdgcn_wmma_f32_16x16x32_bf16(
                    false, a1, false, b0, (short)0, acc[0][1], false, false);
                acc[0][2] = __builtin_amdgcn_wmma_f32_16x16x32_bf16(
                    false, a2, false, b0, (short)0, acc[0][2], false, false);
                acc[0][3] = __builtin_amdgcn_wmma_f32_16x16x32_bf16(
                    false, a3, false, b0, (short)0, acc[0][3], false, false);
                acc[1][0] = __builtin_amdgcn_wmma_f32_16x16x32_bf16(
                    false, a0, false, b1f, (short)0, acc[1][0], false, false);
                acc[1][1] = __builtin_amdgcn_wmma_f32_16x16x32_bf16(
                    false, a1, false, b1f, (short)0, acc[1][1], false, false);
                acc[1][2] = __builtin_amdgcn_wmma_f32_16x16x32_bf16(
                    false, a2, false, b1f, (short)0, acc[1][2], false, false);
                acc[1][3] = __builtin_amdgcn_wmma_f32_16x16x32_bf16(
                    false, a3, false, b1f, (short)0, acc[1][3], false, false);
            }
            float bias0 = (hc == 0) ? b2[(size_t)e * TD + n0] : 0.0f;
            float bias1 = (hc == 0) ? b2[(size_t)e * TD + n1] : 0.0f;
            #pragma unroll
            for (int rt = 0; rt < 4; ++rt)
                #pragma unroll
                for (int v = 0; v < 8; ++v) {
                    int m = rt * 16 + v + 8 * hi;
                    int t = tok_s[m];
                    if (t >= 0) {
                        float w = wt_s[m];
                        atomicAdd(&out[(size_t)t * TD + n0], (acc[0][rt][v] + bias0) * w);
                        atomicAdd(&out[(size_t)t * TD + n1], (acc[1][rt][v] + bias1) * w);
                    }
                }
        }
        __syncthreads();   // hs reused next chunk
    }
}

extern "C" void kernel_launch(void* const* d_in, const int* in_sizes, int n_in,
                              void* d_out, int out_size, void* d_ws, size_t ws_size,
                              hipStream_t stream) {
    const float* x      = (const float*)d_in[0];
    const float* gate_w = (const float*)d_in[1];
    const float* gate_b = (const float*)d_in[2];
    const float* w1     = (const float*)d_in[3];
    const float* b1     = (const float*)d_in[4];
    const float* w2     = (const float*)d_in[5];
    const float* b2     = (const float*)d_in[6];
    float* out = (float*)d_out;

    // workspace layout
    char* ws = (char*)d_ws;
    int*    cnt    = (int*)(ws + 0);                               // 32 B
    int*    tokIdx = (int*)(ws + 256);                             // E*T*4
    float*  tokWt  = (float*)(ws + 256 + (size_t)TE * TT * 4);     // E*T*4
    __bf16* w1T    = (__bf16*)(ws + (size_t)(2u << 20));           // E*D*H*2
    __bf16* w2T    = (__bf16*)(ws + (size_t)(2u << 20) + (size_t)TE * TD * TH * 2);

    hipMemsetAsync(out, 0, (size_t)TT * TD * sizeof(float), stream);
    hipMemsetAsync(cnt, 0, TE * sizeof(int), stream);

    // weights fp32 -> bf16, transposed so both GEMM B-operands are N-major
    moe_transpose_bf16<<<dim3(TH / 32, TD / 32, TE), dim3(32, 8), 0, stream>>>(
        w1, w1T, TD, TH);   // w1T[e][H][D]
    moe_transpose_bf16<<<dim3(TD / 32, TH / 32, TE), dim3(32, 8), 0, stream>>>(
        w2, w2T, TH, TD);   // w2T[e][D][H]

    moe_gate<<<TT / 8, 256, 0, stream>>>(x, gate_w, gate_b, cnt, tokIdx, tokWt);

    // worst case: all tokens on one expert -> T/MT tiles; empty tiles exit fast
    moe_expert_ffn<<<dim3(TT / MT, TE), 256, 0, stream>>>(
        x, b1, b2, w1T, w2T, cnt, tokIdx, tokWt, out);
}
